// geometric_detail_extractor_2241972929093
// MI455X (gfx1250) — matrix operand order repaired and checked by reading, not compiled, not run on hardware
//
#include <hip/hip_runtime.h>
#include <hip/hip_bf16.h>
#include <cstdint>

#define EPSV 1e-5f
#define KNNK 20
#define BB 16
#define NN 2048
#define CC 64

typedef _Float16 v16h __attribute__((ext_vector_type(16)));
typedef _Float16 v8h  __attribute__((ext_vector_type(8)));
typedef float    v8f  __attribute__((ext_vector_type(8)));
typedef uint32_t u32x4 __attribute__((ext_vector_type(4)));
typedef uint32_t u32x8 __attribute__((ext_vector_type(8)));

__device__ __forceinline__ v8f wmma_f16(v16h a, v16h b, v8f c) {
    // D = A(16x32 f16) * B(32x16 f16) + C(16x16 f32)
    return __builtin_amdgcn_wmma_f32_16x16x32_f16(
        /*neg_a=*/false, a, /*neg_b=*/false, b,
        /*c_mod=*/(short)0, c, /*reuse_a=*/false, /*reuse_b=*/false);
}

// Load A-matrix fragment (16x32, f16) for WMMA from LDS weight matrix W (row-major, ldk f16/row).
// Contiguous k order (this code's choice, matching the B fragments): lane L holds row m=L%16; half h=L/16 selects K-run [kbase+16h, +16).
__device__ __forceinline__ v16h load_afrag(const _Float16* W, int ldk, int row, int kbase, int h) {
    const _Float16* base = W + row * ldk + kbase + 16 * h;
    v8h lo = *(const v8h*)(base);
    v8h hi = *(const v8h*)(base + 8);
    v16h a;
    #pragma unroll
    for (int e = 0; e < 8; ++e) { a[e] = lo[e]; a[8 + e] = hi[e]; }
    return a;
}

// Packed-f16 leaky ReLU applied after f32->f16 conversion: pos + 0.2*neg (v_pk_* ops).
__device__ __forceinline__ v8h lrelu_cvt(v8f a) {
    v8h x;
    #pragma unroll
    for (int r = 0; r < 8; ++r) x[r] = (_Float16)a[r];
    v8h zero, slope;
    #pragma unroll
    for (int r = 0; r < 8; ++r) { zero[r] = (_Float16)0.0f; slope[r] = (_Float16)0.2f; }
    v8h pos = __builtin_elementwise_max(x, zero);
    v8h neg = __builtin_elementwise_min(x, zero);
    return pos + slope * neg;
}

// ---------------------------------------------------------------------------
// Tensor Data Mover: async 1D copy of `nelem` f32 from global -> LDS.
// D# built per CDNA5 ISA 8.3/8.4 (group0 + group1, 2-group form, no gather).
// ---------------------------------------------------------------------------
__device__ __forceinline__ void tdm_load_f32(uint32_t lds_off, const float* gsrc, uint32_t nelem) {
    const uint64_t ga = (uint64_t)(uintptr_t)gsrc;
    u32x4 g0;
    g0[0] = 1u;                                    // count=1 (valid), is_restore=0, gather_mode=0
    g0[1] = lds_off;                               // lds_addr [63:32]
    g0[2] = (uint32_t)(ga & 0xffffffffu);          // global_addr[31:0]  -> bits 95:64
    g0[3] = (uint32_t)((ga >> 32) & 0x01ffffffu)   // global_addr[56:32] -> bits 120:96
          | (2u << 30);                            // type=2 ("image")   -> bits 127:126
    u32x8 g1;
    g1[0] = (2u << 16);                            // wg_mask=0, data_size=2 (4 bytes)
    g1[1] = (nelem & 0xffffu) << 16;               // tensor_dim0[15:0]  -> bits 63:48
    g1[2] = ((nelem >> 16) & 0xffffu)              // tensor_dim0[31:16] -> bits 79:64
          | (1u << 16);                            // tensor_dim1 = 1    -> bits 95:80
    g1[3] = (nelem & 0xffffu) << 16;               // tile_dim0          -> bits 127:112
    g1[4] = 1u;                                    // tile_dim1 = 1, tile_dim2 = 0
    g1[5] = nelem;                                 // tensor_dim0_stride[31:0]
    g1[6] = 0u;                                    // stride0[47:32]=0, stride1 lo=0
    g1[7] = 0u;                                    // stride1 hi=0
    asm volatile("tensor_load_to_lds %0, %1" :: "s"(g0), "s"(g1) : "memory");
}

// ---------------------------------------------------------------------------
// Kernel 1: brute-force KNN (top-20 smallest squared distance, incl. self).
// One thread per (b, n). Candidates staged through LDS in chunks of 256.
// ---------------------------------------------------------------------------
__global__ __launch_bounds__(256) void knn_kernel(const float* __restrict__ xyz,
                                                  int* __restrict__ knn_idx) {
    const int t = blockIdx.x * 256 + threadIdx.x;   // t = b*2048 + n (block spans one b)
    const int b = t >> 11;
    const int n = t & (NN - 1);
    const float* xb = xyz + (size_t)b * 3 * NN;
    const float x0 = xb[n], y0 = xb[NN + n], z0 = xb[2 * NN + n];

    float dist[KNNK];
    int   id[KNNK];
    #pragma unroll
    for (int k = 0; k < KNNK; ++k) { dist[k] = 3.0e38f; id[k] = 0; }
    float worst = 3.0e38f;
    int   wslot = 0;

    __shared__ float sx[256], sy[256], sz[256];
    for (int base = 0; base < NN; base += 256) {
        __syncthreads();
        const int jj = base + threadIdx.x;
        sx[threadIdx.x] = xb[jj];
        sy[threadIdx.x] = xb[NN + jj];
        sz[threadIdx.x] = xb[2 * NN + jj];
        __syncthreads();
        for (int q = 0; q < 256; ++q) {
            const float dx = sx[q] - x0, dy = sy[q] - y0, dz = sz[q] - z0;
            const float d = dx * dx + dy * dy + dz * dz;
            if (d < worst) {
                dist[wslot] = d;
                id[wslot] = base + q;
                worst = dist[0]; wslot = 0;
                for (int r = 1; r < KNNK; ++r)
                    if (dist[r] > worst) { worst = dist[r]; wslot = r; }
            }
        }
    }
    int* outp = knn_idx + (size_t)t * KNNK;
    #pragma unroll
    for (int k = 0; k < KNNK; ++k) outp[k] = id[k];
}

// ---------------------------------------------------------------------------
// Kernel 2: fused EdgeConv MLP chain via WMMA.
// Weights staged via TDM (tensor_load_to_lds), BN folded into weights/C-init.
// One wave per point n; K=20 padded to 2 tiles of 16 k-slots; max over K via
// shfl_xor; output written with the UP=2 reshape applied.
// ---------------------------------------------------------------------------
__global__ __launch_bounds__(256) void edgeconv_kernel(
    const float* __restrict__ xf, const int* __restrict__ knn_idx,
    const float* __restrict__ W1, const float* __restrict__ g1, const float* __restrict__ b1,
    const float* __restrict__ m1, const float* __restrict__ v1,
    const float* __restrict__ W2, const float* __restrict__ g2, const float* __restrict__ b2,
    const float* __restrict__ m2, const float* __restrict__ v2,
    const float* __restrict__ W3, float* __restrict__ out, int nwavesTotal) {

    __shared__ __align__(16) float sStage1[32 * 128];   // TDM landing zone: W1 f32
    __shared__ __align__(16) float sStage2[32 * 32];    // TDM landing zone: W2 f32
    __shared__ __align__(16) float sStage3[64 * 32];    // TDM landing zone: W3 f32
    __shared__ __align__(16) _Float16 sW1[32 * 128];    // scale1-folded, f16
    __shared__ __align__(16) _Float16 sW2[32 * 32];     // scale2-folded, f16
    __shared__ __align__(16) _Float16 sW3[64 * 32];     // f16
    __shared__ float sSh1[32];
    __shared__ float sSh2[32];
    __shared__ __align__(32) _Float16 sAct[8][16][32];  // per-wave inter-layer scratch

    const int tid = threadIdx.x;

    // Wave 0 kicks off three async TDM copies of the weights into LDS.
    if (tid < 32) {
        tdm_load_f32((uint32_t)(uintptr_t)&sStage1[0], W1, 32 * 128);
        tdm_load_f32((uint32_t)(uintptr_t)&sStage2[0], W2, 32 * 32);
        tdm_load_f32((uint32_t)(uintptr_t)&sStage3[0], W3, 64 * 32);
        __builtin_amdgcn_s_wait_tensorcnt(0);
    }
    __syncthreads();

    // Cooperative conversion: fold BN scale, f32 -> f16 (reads LDS staging).
    for (int i = tid; i < 32 * 128; i += 256) {
        const int oc = i >> 7;
        const float sc = g1[oc] * rsqrtf(v1[oc] + EPSV);
        sW1[i] = (_Float16)(sStage1[i] * sc);
    }
    for (int i = tid; i < 32 * 32; i += 256) {
        const int oc = i >> 5;
        const float sc = g2[oc] * rsqrtf(v2[oc] + EPSV);
        sW2[i] = (_Float16)(sStage2[i] * sc);
    }
    for (int i = tid; i < 64 * 32; i += 256) sW3[i] = (_Float16)sStage3[i];
    if (tid < 32) {
        const float sc1 = g1[tid] * rsqrtf(v1[tid] + EPSV);
        sSh1[tid] = b1[tid] - m1[tid] * sc1;
        const float sc2 = g2[tid] * rsqrtf(v2[tid] + EPSV);
        sSh2[tid] = b2[tid] - m2[tid] * sc2;
    }
    __syncthreads();

    const int wv   = tid >> 5;
    const int lane = tid & 31;
    const int h    = lane >> 4;   // lane half (selects K-run / M-high)
    const int p    = lane & 15;   // position slot / A-row within tile

    // Persistent A fragments for layers 2 & 3 (reused every iteration).
    v16h a2[2], a3[4];
    #pragma unroll
    for (int t = 0; t < 2; ++t) a2[t] = load_afrag(sW2, 32, 16 * t + p, 0, h);
    #pragma unroll
    for (int t = 0; t < 4; ++t) a3[t] = load_afrag(sW3, 32, 16 * t + p, 0, h);

    // Per-lane BN shifts in C/D layout: value for row m = 16t + 8h + r.
    float sh1r[2][8], sh2r[2][8];
    #pragma unroll
    for (int t = 0; t < 2; ++t)
        #pragma unroll
        for (int r = 0; r < 8; ++r) {
            sh1r[t][r] = sSh1[16 * t + 8 * h + r];
            sh2r[t][r] = sSh2[16 * t + 8 * h + r];
        }

    _Float16* myAct = &sAct[wv][0][0];

    const int gwave = blockIdx.x * 8 + wv;
    for (int g = gwave; g < BB * NN; g += nwavesTotal) {
        const int b = g >> 11;
        const int n = g & (NN - 1);
        const float* xb = xf + (size_t)b * CC * NN;
        const int* idxn = knn_idx + (size_t)g * KNNK;

        float mx[4][8];
        #pragma unroll
        for (int t = 0; t < 4; ++t)
            #pragma unroll
            for (int r = 0; r < 8; ++r) mx[t][r] = -3.0e38f;

        #pragma unroll
        for (int tile = 0; tile < 2; ++tile) {
            const int ks = tile * 16 + p;
            const int kk = ks > (KNNK - 1) ? (KNNK - 1) : ks;   // pad by duplicating: max-safe
            const int j = idxn[kk];

            // ---------------- layer 1: 32 <- 128, K-steps of 32 ----------------
            v8f acc0, acc1;
            #pragma unroll
            for (int r = 0; r < 8; ++r) { acc0[r] = sh1r[0][r]; acc1[r] = sh1r[1][r]; }

            #pragma unroll
            for (int s = 0; s < 4; ++s) {
                // B fragment (32x16 f16): lane column n=p, element e -> K = 16h + e.
                v16h bf;
                const int cb = s * 32;
                #pragma unroll
                for (int e = 0; e < 16; ++e) {
                    const int c = cb + 16 * h + e;
                    float v;
                    if (cb < 64) {
                        v = xb[c * NN + j] - xb[c * NN + n];       // edge diff channels
                    } else {
                        v = xb[(c - 64) * NN + n];                 // center channels
                    }
                    bf[e] = (_Float16)v;
                }
                const v16h a0 = load_afrag(sW1, 128, p, s * 32, h);
                const v16h a1 = load_afrag(sW1, 128, 16 + p, s * 32, h);
                acc0 = wmma_f16(a0, bf, acc0);
                acc1 = wmma_f16(a1, bf, acc1);
            }

            // lrelu (packed f16), relayout via per-wave LDS scratch (wave-ordered DS ops).
            v8h st0 = lrelu_cvt(acc0);
            v8h st1 = lrelu_cvt(acc1);
            *(v8h*)(myAct + p * 32 + 8 * h)      = st0;   // rows 16*0 + 8h..+7
            *(v8h*)(myAct + p * 32 + 16 + 8 * h) = st1;   // rows 16*1 + 8h..+7
            __builtin_amdgcn_wave_barrier();
            v8h blo = *(const v8h*)(myAct + p * 32 + 16 * h);
            v8h bhi = *(const v8h*)(myAct + p * 32 + 16 * h + 8);
            v16h bx;
            #pragma unroll
            for (int e = 0; e < 8; ++e) { bx[e] = blo[e]; bx[8 + e] = bhi[e]; }

            // ---------------- layer 2: 32 <- 32 ----------------
            v8f c20, c21;
            #pragma unroll
            for (int r = 0; r < 8; ++r) { c20[r] = sh2r[0][r]; c21[r] = sh2r[1][r]; }
            c20 = wmma_f16(a2[0], bx, c20);
            c21 = wmma_f16(a2[1], bx, c21);

            st0 = lrelu_cvt(c20);
            st1 = lrelu_cvt(c21);
            __builtin_amdgcn_wave_barrier();
            *(v8h*)(myAct + p * 32 + 8 * h)      = st0;
            *(v8h*)(myAct + p * 32 + 16 + 8 * h) = st1;
            __builtin_amdgcn_wave_barrier();
            blo = *(const v8h*)(myAct + p * 32 + 16 * h);
            bhi = *(const v8h*)(myAct + p * 32 + 16 * h + 8);
            v16h bx2;
            #pragma unroll
            for (int e = 0; e < 8; ++e) { bx2[e] = blo[e]; bx2[8 + e] = bhi[e]; }

            // ---------------- layer 3: 64 <- 32, then running max over k ----------------
            #pragma unroll
            for (int t = 0; t < 4; ++t) {
                v8f c3;
                #pragma unroll
                for (int r = 0; r < 8; ++r) c3[r] = 0.f;
                c3 = wmma_f16(a3[t], bx2, c3);
                #pragma unroll
                for (int r = 0; r < 8; ++r) mx[t][r] = fmaxf(mx[t][r], c3[r]);
            }
        } // tile

        // Max across the 16 k-slots (lanes within each 16-lane half).
        #pragma unroll
        for (int t = 0; t < 4; ++t)
            #pragma unroll
            for (int r = 0; r < 8; ++r) {
                float v = mx[t][r];
                v = fmaxf(v, __shfl_xor(v, 1));
                v = fmaxf(v, __shfl_xor(v, 2));
                v = fmaxf(v, __shfl_xor(v, 4));
                v = fmaxf(v, __shfl_xor(v, 8));
                mx[t][r] = v;
            }

        // out[b, c, 2n+u] = max_k f[b, u*32+c, n]   (UP=2 reshape)
        if (p == 0) {
            #pragma unroll
            for (int t = 0; t < 4; ++t)
                #pragma unroll
                for (int r = 0; r < 8; ++r) {
                    const int m = 16 * t + 8 * h + r;
                    const int c = m & 31;
                    const int u = m >> 5;
                    out[((size_t)b * 32 + c) * (NN * 2) + 2 * n + u] = mx[t][r];
                }
        }
    }
}

extern "C" void kernel_launch(void* const* d_in, const int* in_sizes, int n_in,
                              void* d_out, int out_size, void* d_ws, size_t ws_size,
                              hipStream_t stream) {
    const float* xyz   = (const float*)d_in[0];
    const float* xyz_f = (const float*)d_in[1];
    const float* W1 = (const float*)d_in[2];
    const float* g1 = (const float*)d_in[3];
    const float* b1 = (const float*)d_in[4];
    const float* m1 = (const float*)d_in[5];
    const float* v1 = (const float*)d_in[6];
    const float* W2 = (const float*)d_in[7];
    const float* g2 = (const float*)d_in[8];
    const float* b2 = (const float*)d_in[9];
    const float* m2 = (const float*)d_in[10];
    const float* v2 = (const float*)d_in[11];
    const float* W3 = (const float*)d_in[12];
    float* out = (float*)d_out;

    int* knn_idx = (int*)d_ws;   // B*N*K ints = 2.62 MB of scratch

    knn_kernel<<<(BB * NN) / 256, 256, 0, stream>>>(xyz, knn_idx);

    const int blocks = 1024;                 // 8 waves/block -> 8192 waves, 4 points each
    const int nwaves = blocks * 8;
    edgeconv_kernel<<<blocks, 256, 0, stream>>>(xyz_f, knn_idx,
                                                W1, g1, b1, m1, v1,
                                                W2, g2, b2, m2, v2,
                                                W3, out, nwaves);
}
